// RecurrentBlock_27401891349114
// MI455X (gfx1250) — compile-verified, matrix operand order checked by reading, not run on hardware
//
#include <hip/hip_runtime.h>
#include <hip/hip_bf16.h>
#include <cstdint>
#include <cstddef>

#define B_  4
#define T_  2048
#define D_  2048
#define L_  2048
#define H_  16
#define HD_ 128
#define CW_ 4
#define NC_ 16    // scan chunks per sequence
#define TC_ 128   // timesteps per chunk (NC_*TC_ == T_)
#define SA  40    // LDS row stride in halfs (32 + 8 pad)

typedef __attribute__((ext_vector_type(16))) __bf16 v16bf;
typedef __attribute__((ext_vector_type(8)))  float  v8f;
typedef unsigned short u16;
typedef unsigned int   u32;
typedef unsigned long long u64;

union FragU { u32 u[8]; v16bf v; };

__device__ __forceinline__ u16 f2bf(float f) {
  u32 u = __float_as_uint(f);
  u += 0x7fffu + ((u >> 16) & 1u);   // round-to-nearest-even
  return (u16)(u >> 16);
}

// Load one 16x32 bf16 A-style fragment from LDS (row-major, stride rs halfs).
// CDNA5 16-bit A layout: lane<16 holds K=0..7,16..23 of row (lane&15);
// lane>=16 holds K=8..15,24..31.  Two b128 LDS reads per lane.
__device__ __forceinline__ v16bf ldfrag(const u16* base, int row, int rs, int khalf) {
  FragU f;
  const u16* p = base + row * rs + khalf;
  *(uint4*)(&f.u[0]) = *(const uint4*)(p);       // K = khalf .. khalf+7
  *(uint4*)(&f.u[4]) = *(const uint4*)(p + 16);  // K = 16+khalf .. 23+khalf
  return f.v;
}

// Transposed B staging: thread owns a 2(k) x 8(n) micro-tile; pack the two
// k-halves per n into one dword -> 8 x ds_store_b32 instead of 16 x b16.
__device__ __forceinline__ void packB(u16* Bsb, int bq, int bs,
                                      const uint4& r0, const uint4& r1) {
  const u32* a0 = (const u32*)&r0;
  const u32* a1 = (const u32*)&r1;
  #pragma unroll
  for (int j = 0; j < 8; j++) {
    const u32 lo = (a0[j >> 1] >> ((j & 1) * 16)) & 0xffffu;
    const u32 hi = (a1[j >> 1] >> ((j & 1) * 16)) & 0xffffu;
    *(u32*)&Bsb[(8 * bs + j) * SA + 2 * bq] = lo | (hi << 16);
  }
}

// ---------------------------------------------------------------------------
// Generic bf16 GEMM:  C[M,N] = A[M,K] * B[K,N] + bias, f32 out.
// MODE 0: plain.  MODE 1: tanh-approx GELU.
// 256 thr (8 waves), tile 128x128, BK=32, double-buffered LDS.
// A tile staged with async global->LDS DMA (ASYNCcnt); B staged transposed.
// ---------------------------------------------------------------------------
template <int MODE>
__global__ __launch_bounds__(256) void gemm_bf16(
    const u16* __restrict__ A, const u16* __restrict__ Bw,
    const float* __restrict__ bias, float* __restrict__ C,
    int M, int N, int K) {
  __shared__ u16 As[2][128 * SA];
  __shared__ u16 Bs[2][128 * SA];   // transposed: Bs[n][k]
  const int tid  = threadIdx.x;
  const int lane = tid & 31;
  const int wave = tid >> 5;
  const int wm = wave & 1;             // 2 waves in M
  const int wn = wave >> 1;            // 4 waves in N
  const int m0 = blockIdx.y * 128;
  const int n0 = blockIdx.x * 128;
  const int lrow  = lane & 15;
  const int khalf = (lane & 16) ? 8 : 0;

  const v8f vzero = {0.f,0.f,0.f,0.f,0.f,0.f,0.f,0.f};
  v8f acc[4][2];
  #pragma unroll
  for (int i = 0; i < 4; i++)
    #pragma unroll
    for (int j = 0; j < 2; j++) acc[i][j] = vzero;

  const int ar = tid >> 1;             // A stage: row 0..127
  const int ac = (tid & 1) * 16;       // col half (16 halfs = 32B)
  const int bq = tid >> 4;             // B stage: k-pair 0..15
  const int bs = tid & 15;             // B stage: 8-col group 0..15

  // async DMA of this thread's 32B A slice into LDS buffer `buf`
  auto stageA = [&](int kb, int buf) {
    const u64 g  = (u64)(A + (size_t)(m0 + ar) * K + kb + ac);
    const u32 l0 = (u32)(size_t)&As[buf][ar * SA + ac];
    asm volatile("global_load_async_to_lds_b128 %0, %1, off"
                 :: "v"(l0), "v"(g) : "memory");
    asm volatile("global_load_async_to_lds_b128 %0, %1, off offset:16"
                 :: "v"(l0), "v"(g) : "memory");
  };
  auto loadB = [&](int kb, uint4& r0, uint4& r1) {
    const u16* g = Bw + (size_t)(kb + 2 * bq) * N + n0 + 8 * bs;
    r0 = *(const uint4*)g;
    r1 = *(const uint4*)(g + N);
  };

  uint4 rb0, rb1;
  stageA(0, 0);
  loadB(0, rb0, rb1);
  packB(&Bs[0][0], bq, bs, rb0, rb1);
  asm volatile("s_wait_asynccnt 0x0" ::: "memory");
  __syncthreads();

  int cur = 0;
  for (int kb = 0; kb < K; kb += 32) {
    const bool nxt = (kb + 32) < K;
    if (nxt) {                       // prefetch next tile (overlaps WMMAs)
      stageA(kb + 32, cur ^ 1);
      loadB(kb + 32, rb0, rb1);
    }
    v16bf af[4], bfr[2];
    #pragma unroll
    for (int mi = 0; mi < 4; mi++)
      af[mi] = ldfrag(&As[cur][0], wm * 64 + mi * 16 + lrow, SA, khalf);
    #pragma unroll
    for (int ni = 0; ni < 2; ni++)
      bfr[ni] = ldfrag(&Bs[cur][0], wn * 32 + ni * 16 + lrow, SA, khalf);
    #pragma unroll
    for (int mi = 0; mi < 4; mi++)
      #pragma unroll
      for (int ni = 0; ni < 2; ni++)
        acc[mi][ni] = __builtin_amdgcn_wmma_f32_16x16x32_bf16(
            false, af[mi], false, bfr[ni], (short)0, acc[mi][ni], false, false);
    if (nxt) {
      packB(&Bs[cur ^ 1][0], bq, bs, rb0, rb1);
      asm volatile("s_wait_asynccnt 0x0" ::: "memory");
    }
    __syncthreads();
    cur ^= 1;
  }

  const int mh = (lane & 16) ? 8 : 0;  // C layout: lanes>=16 hold M=8..15
  #pragma unroll
  for (int mi = 0; mi < 4; mi++) {
    #pragma unroll
    for (int ni = 0; ni < 2; ni++) {
      const int gn = n0 + wn * 32 + ni * 16 + lrow;
      const float bv = bias[gn];
      #pragma unroll
      for (int j = 0; j < 8; j++) {
        const int gm = m0 + wm * 64 + mi * 16 + mh + j;
        float v = acc[mi][ni][j] + bv;
        if (MODE == 1) {
          const float x3 = v * v * v;
          v = 0.5f * v * (1.f + tanhf(0.7978845608028654f * (v + 0.044715f * x3)));
        }
        C[(size_t)gm * N + gn] = v;
      }
    }
  }
}

// ---------------------------------------------------------------------------
// Fused gate kernel: per head h, gate_x = sigmoid(xh@Wgx+b), gate_a likewise,
// then full RG-LRU pointwise epilogue producing a[t,l] and normed[t,l].
// Block tile 64(M) x 128(N=head width), K=128, 8 waves of 32x32.
// NOTE: n_out aliases xc (read-before-write per element, same thread).
// ---------------------------------------------------------------------------
__global__ __launch_bounds__(256) void gate_kernel(
    const u16* __restrict__ Axc,   // xc in bf16 [B*T, L]
    const u16* __restrict__ Wgx,   // [H,128,128] bf16
    const u16* __restrict__ Wga,
    const float* __restrict__ bgx, const float* __restrict__ bga,
    const float* __restrict__ sp,  // softplus(a_param) [L]
    const float* xc,               // f32 [B*T, L]   (aliases n_out!)
    float* a_out, float* n_out) {
  __shared__ u16 As[64 * SA];
  __shared__ u16 Bxs[128 * SA];
  __shared__ u16 Bas[128 * SA];
  const int h  = blockIdx.x;
  const int m0 = blockIdx.y * 64;
  const int tid = threadIdx.x, lane = tid & 31, wave = tid >> 5;
  const int wm = wave & 1, wn = wave >> 1;
  const int lrow  = lane & 15;
  const int khalf = (lane & 16) ? 8 : 0;

  const v8f vzero = {0.f,0.f,0.f,0.f,0.f,0.f,0.f,0.f};
  v8f ax[2][2], aa[2][2];
  #pragma unroll
  for (int i = 0; i < 2; i++)
    #pragma unroll
    for (int j = 0; j < 2; j++) { ax[i][j] = vzero; aa[i][j] = vzero; }

  const int ar = tid >> 2;           // 0..63
  const int ac = (tid & 3) * 8;      // 0..24
  const int bq = tid >> 4;           // k-pair 0..15
  const int bs = tid & 15;           // 8-col group 0..15
  const u16* wx = Wgx + h * HD_ * HD_;
  const u16* wa = Wga + h * HD_ * HD_;

  for (int kb = 0; kb < HD_; kb += 32) {
    { const u16* g = Axc + (size_t)(m0 + ar) * L_ + h * HD_ + kb + ac;
      *(uint4*)&As[ar * SA + ac] = *(const uint4*)g; }
    { const u16* g = wx + (size_t)(kb + 2 * bq) * HD_ + 8 * bs;
      uint4 r0 = *(const uint4*)g, r1 = *(const uint4*)(g + HD_);
      packB(Bxs, bq, bs, r0, r1);
      g = wa + (size_t)(kb + 2 * bq) * HD_ + 8 * bs;
      r0 = *(const uint4*)g; r1 = *(const uint4*)(g + HD_);
      packB(Bas, bq, bs, r0, r1);
    }
    __syncthreads();
    v16bf af[2];
    #pragma unroll
    for (int mi = 0; mi < 2; mi++)
      af[mi] = ldfrag(As, wm * 32 + mi * 16 + lrow, SA, khalf);
    #pragma unroll
    for (int ni = 0; ni < 2; ni++) {
      const int nr = wn * 32 + ni * 16 + lrow;
      v16bf bx = ldfrag(Bxs, nr, SA, khalf);
      v16bf ba = ldfrag(Bas, nr, SA, khalf);
      #pragma unroll
      for (int mi = 0; mi < 2; mi++) {
        ax[mi][ni] = __builtin_amdgcn_wmma_f32_16x16x32_bf16(
            false, af[mi], false, bx, (short)0, ax[mi][ni], false, false);
        aa[mi][ni] = __builtin_amdgcn_wmma_f32_16x16x32_bf16(
            false, af[mi], false, ba, (short)0, aa[mi][ni], false, false);
      }
    }
    __syncthreads();
  }

  const int mh = (lane & 16) ? 8 : 0;
  #pragma unroll
  for (int mi = 0; mi < 2; mi++) {
    #pragma unroll
    for (int ni = 0; ni < 2; ni++) {
      const int ln = wn * 32 + ni * 16 + lrow;
      const int l  = h * HD_ + ln;
      const float bxv = bgx[l], bav = bga[l], spv = sp[l];
      #pragma unroll
      for (int j = 0; j < 8; j++) {
        const int r = m0 + wm * 32 + mi * 16 + mh + j;
        const int t = r & (T_ - 1);
        const size_t k = (size_t)r * L_ + l;
        const float gx = 1.f / (1.f + expf(-(ax[mi][ni][j] + bxv)));
        const float ga = 1.f / (1.f + expf(-(aa[mi][ni][j] + bav)));
        const float loga = -8.f * ga * spv;
        float av, mult;
        if (t == 0) { av = 0.f; mult = 1.f; }
        else        { av = expf(loga); mult = sqrtf(fmaxf(1.f - expf(2.f * loga), 0.f)); }
        const float xcv = xc[k];       // read BEFORE aliased write below
        a_out[k] = av;
        n_out[k] = xcv * gx * mult;
      }
    }
  }
}

// ---------------------------------------------------------------------------
// Pointwise / scan kernels
// ---------------------------------------------------------------------------
__global__ void f2bf_kernel(const float* __restrict__ in, u16* __restrict__ out, size_t n) {
  const size_t i = (size_t)blockIdx.x * blockDim.x + threadIdx.x;
  if (i < n) out[i] = f2bf(in[i]);
}

__global__ void softplus_kernel(const float* __restrict__ ap, float* __restrict__ sp, int n) {
  const int i = blockIdx.x * blockDim.x + threadIdx.x;
  if (i < n) { const float v = ap[i]; sp[i] = (v > 20.f) ? v : log1pf(expf(v)); }
}

__global__ void conv_kernel(const float* __restrict__ xp,
                            const float* __restrict__ cw, const float* __restrict__ cb,
                            float* __restrict__ xc, u16* __restrict__ xcbf) {
  const size_t idx = (size_t)blockIdx.x * 256 + threadIdx.x;   // [B*T, L]
  const int l = (int)(idx & (L_ - 1));
  const size_t r = idx >> 11;
  const int t = (int)(r & (T_ - 1));
  float v = cb[l];
  #pragma unroll
  for (int i = 0; i < CW_; i++) {
    const int ts = t - (CW_ - 1) + i;
    if (ts >= 0) v += cw[i * L_ + l] * xp[(r + i - (CW_ - 1)) * (size_t)L_ + l];
  }
  xc[idx] = v;
  xcbf[idx] = f2bf(v);
}

__global__ void scan_chunk(const float* __restrict__ a, const float* __restrict__ nrm,
                           float* __restrict__ Ac, float* __restrict__ Hc) {
  const int idx = blockIdx.x * 256 + threadIdx.x;    // (b, c, l)
  const int l = idx & (L_ - 1);
  const int c = (idx >> 11) & (NC_ - 1);
  const int b = idx >> 15;
  float A = 1.f, Hv = 0.f;
  size_t base = ((size_t)b * T_ + (size_t)c * TC_) * L_ + l;
  for (int i = 0; i < TC_; i++) {
    const float av = a[base], nv = nrm[base];
    Hv = av * Hv + nv;
    A *= av;
    base += L_;
  }
  Ac[idx] = A; Hc[idx] = Hv;
}

__global__ void scan_carry(const float* __restrict__ Ac, const float* __restrict__ Hc,
                           float* __restrict__ Hin) {
  const int idx = blockIdx.x * 256 + threadIdx.x;    // (b, l)
  const int l = idx & (L_ - 1);
  const int b = idx >> 11;
  float carry = 0.f;
  for (int c = 0; c < NC_; c++) {
    const int k = (b * NC_ + c) * L_ + l;
    Hin[k] = carry;
    carry = Ac[k] * carry + Hc[k];
  }
}

__global__ void scan_apply(const float* __restrict__ a, const float* __restrict__ nrm,
                           const float* __restrict__ Hin, const float* __restrict__ y,
                           u16* __restrict__ hy) {
  const int idx = blockIdx.x * 256 + threadIdx.x;
  const int l = idx & (L_ - 1);
  const int c = (idx >> 11) & (NC_ - 1);
  const int b = idx >> 15;
  float h = Hin[idx];
  size_t base = ((size_t)b * T_ + (size_t)c * TC_) * L_ + l;
  for (int i = 0; i < TC_; i++) {
    h = a[base] * h + nrm[base];
    hy[base] = f2bf(h * y[base]);
    base += L_;
  }
}

// ---------------------------------------------------------------------------
extern "C" void kernel_launch(void* const* d_in, const int* in_sizes, int n_in,
                              void* d_out, int out_size, void* d_ws, size_t ws_size,
                              hipStream_t stream) {
  const float* x     = (const float*)d_in[0];
  const float* Wy    = (const float*)d_in[1];
  const float* by    = (const float*)d_in[2];
  const float* Wx    = (const float*)d_in[3];
  const float* bx    = (const float*)d_in[4];
  const float* Wout  = (const float*)d_in[5];
  const float* bout  = (const float*)d_in[6];
  const float* convw = (const float*)d_in[7];
  const float* convb = (const float*)d_in[8];
  const float* gxw   = (const float*)d_in[9];
  const float* gxb   = (const float*)d_in[10];
  const float* gaw   = (const float*)d_in[11];
  const float* gab   = (const float*)d_in[12];
  const float* ap    = (const float*)d_in[13];

  const size_t MT = (size_t)B_ * T_;      // 8192 rows
  const size_t NX = MT * D_;              // 16,777,216 elements

  // Workspace carve (~283 MB total with aliasing)
  char* p = (char*)d_ws;
  auto carve = [&](size_t bytes) { char* r = p; p += (bytes + 255) & ~(size_t)255; return r; };
  u16*   xbf     = (u16*)  carve(NX * 2);            // reused as hy_bf after GEMM1/2
  u16*   wy_bf   = (u16*)  carve((size_t)D_ * L_ * 2);
  u16*   wx_bf   = (u16*)  carve((size_t)D_ * L_ * 2);
  u16*   wout_bf = (u16*)  carve((size_t)L_ * D_ * 2);
  u16*   gwx_bf  = (u16*)  carve((size_t)H_ * HD_ * HD_ * 2);
  u16*   gwa_bf  = (u16*)  carve((size_t)H_ * HD_ * HD_ * 2);
  float* spb     = (float*)carve((size_t)L_ * 4);
  float* ybuf    = (float*)carve(NX * 4);
  float* xpbuf   = (float*)carve(NX * 4);            // reused as a_buf after conv
  float* xcbuf   = (float*)carve(NX * 4);            // reused as normed (same-thread RAW safe)
  u16*   xcbf    = (u16*)  carve(NX * 2);
  float* Ac      = (float*)carve((size_t)B_ * NC_ * L_ * 4);
  float* Hc      = (float*)carve((size_t)B_ * NC_ * L_ * 4);
  float* Hin     = (float*)carve((size_t)B_ * NC_ * L_ * 4);

  // 0) fp32 -> bf16 conversions + softplus(a_param)
  f2bf_kernel<<<(unsigned)((NX + 255) / 256), 256, 0, stream>>>(x, xbf, NX);
  f2bf_kernel<<<(unsigned)(((size_t)D_ * L_ + 255) / 256), 256, 0, stream>>>(Wy, wy_bf, (size_t)D_ * L_);
  f2bf_kernel<<<(unsigned)(((size_t)D_ * L_ + 255) / 256), 256, 0, stream>>>(Wx, wx_bf, (size_t)D_ * L_);
  f2bf_kernel<<<(unsigned)(((size_t)L_ * D_ + 255) / 256), 256, 0, stream>>>(Wout, wout_bf, (size_t)L_ * D_);
  f2bf_kernel<<<(unsigned)(((size_t)H_ * HD_ * HD_ + 255) / 256), 256, 0, stream>>>(gxw, gwx_bf, (size_t)H_ * HD_ * HD_);
  f2bf_kernel<<<(unsigned)(((size_t)H_ * HD_ * HD_ + 255) / 256), 256, 0, stream>>>(gaw, gwa_bf, (size_t)H_ * HD_ * HD_);
  softplus_kernel<<<(L_ + 255) / 256, 256, 0, stream>>>(ap, spb, L_);

  // 1) y = gelu(x@Wy + by);  xp = x@Wx + bx
  dim3 gBig(L_ / 128, (unsigned)(MT / 128));
  gemm_bf16<1><<<gBig, 256, 0, stream>>>(xbf, wy_bf, by, ybuf, (int)MT, L_, D_);
  gemm_bf16<0><<<gBig, 256, 0, stream>>>(xbf, wx_bf, bx, xpbuf, (int)MT, L_, D_);

  // 2) causal depthwise conv -> xc (f32 + bf16)
  conv_kernel<<<(unsigned)(NX / 256), 256, 0, stream>>>(xpbuf, convw, convb, xcbuf, xcbf);

  // 3) fused gate GEMMs + RG-LRU pointwise -> a (in xpbuf), normed (in xcbuf)
  dim3 gGate(H_, (unsigned)(MT / 64));
  gate_kernel<<<gGate, 256, 0, stream>>>(xcbf, gwx_bf, gwa_bf, gxb, gab, spb,
                                         xcbuf, xpbuf, xcbuf);

  // 4) linear recurrence h_t = a_t h_{t-1} + normed_t (3-pass chunked scan),
  //    fusing hy = (h * y) -> bf16 into xbf (reuse)
  scan_chunk<<<(B_ * NC_ * L_) / 256, 256, 0, stream>>>(xpbuf, xcbuf, Ac, Hc);
  scan_carry<<<(B_ * L_) / 256, 256, 0, stream>>>(Ac, Hc, Hin);
  scan_apply<<<(B_ * NC_ * L_) / 256, 256, 0, stream>>>(xpbuf, xcbuf, Hin, ybuf, xbf);

  // 5) out = hy @ Wout + bout
  dim3 gOut(D_ / 128, (unsigned)(MT / 128));
  gemm_bf16<0><<<gOut, 256, 0, stream>>>(xbf, wout_bf, bout, (float*)d_out, (int)MT, D_, L_);
}
